// HyperedgeReadout_52166672777346
// MI455X (gfx1250) — matrix operand order repaired and checked by reading, not compiled, not run on hardware
//
#include <hip/hip_runtime.h>
#include <hip/hip_bf16.h>

// ---------------------------------------------------------------------------
// HyperedgeReadout for MI455X (gfx1250, wave32, WMMA).
// Heavy step: feat(300000x1024) @ W1(1024x256) done with
// v_wmma_f32_16x16x32_bf16, feat staged in LDS in bf16, W1 pre-converted
// to bf16 once (L2-resident, 512KB).
// ---------------------------------------------------------------------------

#define E_EDGES   300000
#define H_DIM     256
#define K_DIM     1024           // 4*H
#define NC_CASES  20000
#define ET        64             // edge tile per block (M dimension)
#define EPSF      1e-6f

typedef __attribute__((ext_vector_type(16))) __bf16 v16bf;
typedef __attribute__((ext_vector_type(8)))  float  v8f;

union BF16x16 {
    uint4 q[2];
    v16bf v;
};

static __device__ __forceinline__ unsigned short f2bf(float f) {
    // round-to-nearest-even fp32 -> bf16
    unsigned u = __float_as_uint(f);
    unsigned r = u + 0x7FFFu + ((u >> 16) & 1u);
    return (unsigned short)(r >> 16);
}

static __device__ __forceinline__ unsigned ordered_enc(float f) {
    unsigned u = __float_as_uint(f);
    return (u & 0x80000000u) ? ~u : (u | 0x80000000u);
}
static __device__ __forceinline__ float ordered_dec(unsigned u) {
    unsigned b = (u & 0x80000000u) ? (u & 0x7FFFFFFFu) : ~u;
    return __uint_as_float(b);
}

// ------------------------- kernel 0: W1 fp32 -> bf16 ------------------------
__global__ void k_convert_w1(const float* __restrict__ W1,
                             unsigned short* __restrict__ W1b) {
    int i = blockIdx.x * blockDim.x + threadIdx.x;   // 1024*256 elements
    if (i < K_DIM * H_DIM) W1b[i] = f2bf(W1[i]);
}

// --------------------- kernel 1: segment sums (deg, sum) --------------------
// one block (256 threads) per edge
__global__ void k_segsum(const float* __restrict__ edge_repr,
                         const float* __restrict__ edge_weight,
                         const int*   __restrict__ case_idx,
                         float* __restrict__ case_deg,
                         float* __restrict__ case_sum) {
    int e = blockIdx.x;
    int h = threadIdx.x;
    int ci = case_idx[e];
    float w = edge_weight[e];
    float er = edge_repr[(size_t)e * H_DIM + h];
    atomicAdd(&case_sum[(size_t)ci * H_DIM + h], w * er);
    if (h == 0) atomicAdd(&case_deg[ci], w);
}

// -------- kernel 2: feat build + bf16 WMMA GEMM + fused W2 -> logits --------
extern __shared__ __align__(16) char smem_raw[];

__global__ void __launch_bounds__(256)
k_gemm_logits(const float* __restrict__ edge_repr,
              const float* __restrict__ edge_weight,
              const int*   __restrict__ case_idx,
              const float* __restrict__ case_deg,
              const float* __restrict__ case_sum,
              const unsigned short* __restrict__ W1b,
              const float* __restrict__ b1g,
              const float* __restrict__ W2g,
              const float* __restrict__ b2g,
              float* __restrict__ logits) {
    unsigned short* feat = (unsigned short*)smem_raw;            // ET x 1024 bf16
    float* llds = (float*)(smem_raw + (size_t)ET * K_DIM * 2);   // ET logits accum

    const int tid = threadIdx.x;
    const int e0  = blockIdx.x * ET;

    if (tid < ET) llds[tid] = 0.0f;

    // ---- phase 1: build feat tile (ET x 1024) in bf16 -----------------------
    // 256 threads cover h = 0..255 for one edge per iteration.
    for (int i = 0; i < ET; ++i) {
        int e = e0 + i;
        float er = 0.0f, ctx = 0.0f;
        if (e < E_EDGES) {
            int   ci  = case_idx[e];
            float w   = edge_weight[e];
            er        = edge_repr[(size_t)e * H_DIM + tid];
            float cs  = case_sum[(size_t)ci * H_DIM + tid];
            float deg = case_deg[ci];
            float weighted  = er * w;
            float remaining = deg - w;
            float base  = cs / fmaxf(deg, EPSF);
            float loocv = (cs - weighted) / fmaxf(remaining, 1e-8f);
            ctx = (remaining <= EPSF) ? base : loocv;
        }
        unsigned short* row = feat + (size_t)i * K_DIM;
        row[tid]           = f2bf(er);
        row[H_DIM + tid]   = f2bf(ctx);
        row[2*H_DIM + tid] = f2bf(er * ctx);
        row[3*H_DIM + tid] = f2bf(fabsf(er - ctx));
    }
    __syncthreads();

    // ---- phase 2: GEMM feat(ETx1024) @ W1b(1024x256), 8 waves over N --------
    const int lane    = tid & 31;
    const int halfsel = lane >> 4;          // 0: lanes 0-15, 1: lanes 16-31
    const int ncol    = lane & 15;
    const int n0      = (tid >> 5) * 32;    // each wave owns 32 columns

    v8f acc[4][2];
#pragma unroll
    for (int m = 0; m < 4; ++m)
#pragma unroll
        for (int nt = 0; nt < 2; ++nt)
            acc[m][nt] = (v8f){0.f,0.f,0.f,0.f,0.f,0.f,0.f,0.f};

    for (int kk = 0; kk < K_DIM; kk += 32) {
        // B fragments: lane holds column N = n0+ncol (+16), K = kb+s (s<8), kb+16+(s-8) (s>=8).
        const unsigned short* wrow = W1b + (size_t)(kk + halfsel * 8) * H_DIM + n0 + ncol;
        BF16x16 bf0, bf1;
#pragma unroll
        for (int s = 0; s < 16; ++s) {
            int ks = (s < 8) ? s : s + 8;
            ((unsigned short*)&bf0)[s] = wrow[(size_t)ks * H_DIM];
            ((unsigned short*)&bf1)[s] = wrow[(size_t)ks * H_DIM + 16];
        }

        const int kb = kk + halfsel * 8;    // A-fragment K base for this half
#pragma unroll
        for (int m = 0; m < 4; ++m) {
            const unsigned short* arow =
                feat + (size_t)(m * 16 + (lane & 15)) * K_DIM + kb;
            BF16x16 a;
            a.q[0] = *(const uint4*)(arow);        // K = kb .. kb+7
            a.q[1] = *(const uint4*)(arow + 16);   // K = kb+16 .. kb+23
            acc[m][0] = __builtin_amdgcn_wmma_f32_16x16x32_bf16(
                false, a.v, false, bf0.v, (short)0, acc[m][0], false, false);
            acc[m][1] = __builtin_amdgcn_wmma_f32_16x16x32_bf16(
                false, a.v, false, bf1.v, (short)0, acc[m][1], false, false);
        }
    }

    // ---- epilogue: bias + relu + fused (h @ W2) reduction into LDS ----------
#pragma unroll
    for (int m = 0; m < 4; ++m) {
        float p[8];
#pragma unroll
        for (int r = 0; r < 8; ++r) p[r] = 0.0f;
#pragma unroll
        for (int nt = 0; nt < 2; ++nt) {
            int N = n0 + nt * 16 + ncol;
            float bb = b1g[N];
            float ww = W2g[N];
#pragma unroll
            for (int r = 0; r < 8; ++r) {
                float v = acc[m][nt][r] + bb;
                v = v > 0.0f ? v : 0.0f;
                p[r] += v * ww;
            }
        }
#pragma unroll
        for (int r = 0; r < 8; ++r) {
            int row = m * 16 + r + 8 * halfsel;   // local edge index
            atomicAdd(&llds[row], p[r]);
        }
    }
    __syncthreads();

    if (tid < ET) {
        int e = e0 + tid;
        if (e < E_EDGES) logits[e] = llds[tid] + b2g[0];
    }
}

// ----------------- kernel 3: segment max (ordered-uint trick) ---------------
__global__ void k_gmax(const float* __restrict__ logits,
                       const int*   __restrict__ case_idx,
                       unsigned*    __restrict__ gmax_u) {
    int e = blockIdx.x * blockDim.x + threadIdx.x;
    if (e >= E_EDGES) return;
    atomicMax(&gmax_u[case_idx[e]], ordered_enc(logits[e]));
}

// ------------------- kernel 4: exp + segment sum of exp ---------------------
__global__ void k_expsum(const float* __restrict__ logits,
                         const int*   __restrict__ case_idx,
                         const unsigned* __restrict__ gmax_u,
                         float* __restrict__ gsum,
                         float* __restrict__ expv) {
    int e = blockIdx.x * blockDim.x + threadIdx.x;
    if (e >= E_EDGES) return;
    int ci = case_idx[e];
    float g = ordered_dec(gmax_u[ci]);
    float ev = __expf(logits[e] - g);
    expv[e] = ev;
    atomicAdd(&gsum[ci], ev);
}

// ------------------ kernel 5: final weighted segment sum --------------------
// one block (256 threads) per edge
__global__ void k_final(const float* __restrict__ edge_repr,
                        const float* __restrict__ edge_weight,
                        const int*   __restrict__ case_idx,
                        const float* __restrict__ case_deg,
                        const float* __restrict__ gsum,
                        const float* __restrict__ expv,
                        float* __restrict__ out) {
    int e = blockIdx.x;
    int h = threadIdx.x;
    int ci = case_idx[e];
    float alpha   = expv[e] / (gsum[ci] + 1e-8f);
    float uniform = edge_weight[e] / (case_deg[ci] + 1e-8f);
    float fa = 0.8f * alpha + 0.2f * uniform;
    float er = edge_repr[(size_t)e * H_DIM + h];
    atomicAdd(&out[(size_t)ci * H_DIM + h], fa * er);
}

// ---------------------------------------------------------------------------
extern "C" void kernel_launch(void* const* d_in, const int* in_sizes, int n_in,
                              void* d_out, int out_size, void* d_ws, size_t ws_size,
                              hipStream_t stream) {
    const float* edge_repr   = (const float*)d_in[0];
    const float* edge_weight = (const float*)d_in[1];
    const int*   case_idx    = (const int*)d_in[2];
    // d_in[3] = num_case scalar (compile-time constant NC_CASES used)
    const float* W1 = (const float*)d_in[4];
    const float* b1 = (const float*)d_in[5];
    const float* W2 = (const float*)d_in[6];
    const float* b2 = (const float*)d_in[7];
    float* out = (float*)d_out;

    // workspace partition (256B aligned regions)
    char* ws = (char*)d_ws;
    size_t off = 0;
    auto take = [&](size_t bytes) -> char* {
        char* p = ws + off;
        off = (off + bytes + 255) & ~(size_t)255;
        return p;
    };
    unsigned short* W1b   = (unsigned short*)take((size_t)K_DIM * H_DIM * 2);
    float* case_deg       = (float*)take((size_t)NC_CASES * 4);
    float* case_sum       = (float*)take((size_t)NC_CASES * H_DIM * 4);
    float* logits         = (float*)take((size_t)E_EDGES * 4);
    unsigned* gmax_u      = (unsigned*)take((size_t)NC_CASES * 4);
    float* gsum           = (float*)take((size_t)NC_CASES * 4);
    float* expv           = (float*)take((size_t)E_EDGES * 4);

    // zero accumulators + output
    hipMemsetAsync(case_deg, 0, (size_t)NC_CASES * 4, stream);
    hipMemsetAsync(case_sum, 0, (size_t)NC_CASES * H_DIM * 4, stream);
    hipMemsetAsync(gmax_u,   0, (size_t)NC_CASES * 4, stream);
    hipMemsetAsync(gsum,     0, (size_t)NC_CASES * 4, stream);
    hipMemsetAsync(out,      0, (size_t)out_size * 4, stream);

    // 0) W1 -> bf16
    k_convert_w1<<<(K_DIM * H_DIM + 255) / 256, 256, 0, stream>>>(W1, W1b);

    // 1) segment sums
    k_segsum<<<E_EDGES, H_DIM, 0, stream>>>(edge_repr, edge_weight, case_idx,
                                            case_deg, case_sum);

    // 2) feat + WMMA GEMM + fused W2 -> logits
    {
        int nblocks = (E_EDGES + ET - 1) / ET;
        size_t smem = (size_t)ET * K_DIM * 2 + (size_t)ET * 4;  // 128KB + 256B
        k_gemm_logits<<<nblocks, 256, smem, stream>>>(
            edge_repr, edge_weight, case_idx, case_deg, case_sum,
            W1b, b1, W2, b2, logits);
    }

    // 3) segment max
    k_gmax<<<(E_EDGES + 255) / 256, 256, 0, stream>>>(logits, case_idx, gmax_u);

    // 4) exp + segment sum
    k_expsum<<<(E_EDGES + 255) / 256, 256, 0, stream>>>(logits, case_idx,
                                                        gmax_u, gsum, expv);

    // 5) final weighted readout
    k_final<<<E_EDGES, H_DIM, 0, stream>>>(edge_repr, edge_weight, case_idx,
                                           case_deg, gsum, expv, out);
}
